// CandidateAware_ClickedNewsAttention_54236847014274
// MI455X (gfx1250) — matrix operand order repaired and checked by reading, not compiled, not run on hardware
//
#include <hip/hip_runtime.h>
#include <hip/hip_bf16.h>

typedef __bf16 v16bf __attribute__((ext_vector_type(16)));
typedef float  v8f   __attribute__((ext_vector_type(8)));
typedef unsigned int u32x4 __attribute__((ext_vector_type(4)));

#define B_  256
#define H_  200
#define N_  64
#define D_  640
#define DT_ 256
#define NH_ 10
#define HD_ 64

union Frag {
    v16bf v;
    u32x4 q[2];
};

__device__ __forceinline__ unsigned short f2bf(float f) {
    unsigned int u = __float_as_uint(f);
    u += 0x7FFFu + ((u >> 16) & 1u);
    return (unsigned short)(u >> 16);
}

__global__ __launch_bounds__(256)
void cvt_f32_bf16(const float* __restrict__ src, unsigned short* __restrict__ dst, int n) {
    int i = blockIdx.x * blockDim.x + threadIdx.x;
    if (i < n) dst[i] = f2bf(src[i]);
}

// ---------------- Q projection: [B*N,256] x Wq^T[256->640] + bq ----------------
// outputs Q_bf (bf16) and per-row L2 norm (for query softmax weights)
__global__ __launch_bounds__(256)
void qproj_kernel(const float* __restrict__ X,
                  const unsigned short* __restrict__ Wb,
                  const float* __restrict__ bias,
                  unsigned short* __restrict__ Qbf,
                  float* __restrict__ rnorm) {
    __shared__ unsigned short xb[16][DT_];   // 8 KB
    __shared__ float accb[16][D_];           // 40 KB
    __shared__ float redS[16][16];
    const int m0 = blockIdx.x * 16;
    const int tid = threadIdx.x, lane = tid & 31, wave = tid >> 5;
    const int l16 = lane & 15;
    const int kfirst = (lane < 16) ? 0 : 8;
    const int bsel = (lane >> 4) << 3;
    const int rbase = (lane >> 4) << 3;

    for (int e = tid; e < 16 * DT_; e += 256) {
        int r = e >> 8, c = e & 255;
        xb[r][c] = f2bf(X[(size_t)(m0 + r) * DT_ + c]);
    }
    __syncthreads();

    v8f acc[5];
    for (int t = 0; t < 5; ++t) acc[t] = (v8f){0.f,0.f,0.f,0.f,0.f,0.f,0.f,0.f};

    for (int ks = 0; ks < DT_ / 32; ++ks) {
        Frag a;
        a.q[0] = *(const u32x4*)&xb[l16][ks * 32 + kfirst];
        a.q[1] = *(const u32x4*)&xb[l16][ks * 32 + kfirst + 16];
        for (int t = 0; t < 5; ++t) {
            int coln = wave * 80 + t * 16 + l16;
            const unsigned short* wp = &Wb[(size_t)coln * DT_ + ks * 32 + bsel];
            Frag b;
            b.q[0] = *(const u32x4*)(wp);
            b.q[1] = *(const u32x4*)(wp + 16);
            acc[t] = __builtin_amdgcn_wmma_f32_16x16x32_bf16(
                false, a.v, false, b.v, (short)0, acc[t], false, false);
        }
    }
    for (int t = 0; t < 5; ++t)
        for (int r = 0; r < 8; ++r)
            accb[rbase + r][wave * 80 + t * 16 + l16] = acc[t][r];
    __syncthreads();

    {   // bias, bf16 store, partial sum-of-squares
        int r = tid >> 4, seg = tid & 15;
        float sq = 0.f;
        for (int c = seg * 40; c < seg * 40 + 40; ++c) {
            float v = accb[r][c] + bias[c];
            Qbf[(size_t)(m0 + r) * D_ + c] = f2bf(v);
            sq += v * v;
        }
        redS[r][seg] = sq;
    }
    __syncthreads();
    if ((tid & 15) == 0) {
        int r = tid >> 4;
        float s = 0.f;
        for (int i = 0; i < 16; ++i) s += redS[r][i];
        rnorm[m0 + r] = sqrtf(s);
    }
}

// ---------------- K projection: [B*H,256] x Wk^T[256->640] + bk ----------------
__global__ __launch_bounds__(256)
void kproj_kernel(const float* __restrict__ X,
                  const unsigned short* __restrict__ Wb,
                  const float* __restrict__ bias,
                  unsigned short* __restrict__ Kbf) {
    __shared__ unsigned short xb[16][DT_];
    const int m0 = blockIdx.x * 16;
    const int tid = threadIdx.x, lane = tid & 31, wave = tid >> 5;
    const int l16 = lane & 15;
    const int kfirst = (lane < 16) ? 0 : 8;
    const int bsel = (lane >> 4) << 3;
    const int rbase = (lane >> 4) << 3;

    for (int e = tid; e < 16 * DT_; e += 256) {
        int r = e >> 8, c = e & 255;
        xb[r][c] = f2bf(X[(size_t)(m0 + r) * DT_ + c]);
    }
    __syncthreads();

    v8f acc[5];
    for (int t = 0; t < 5; ++t) acc[t] = (v8f){0.f,0.f,0.f,0.f,0.f,0.f,0.f,0.f};

    for (int ks = 0; ks < DT_ / 32; ++ks) {
        Frag a;
        a.q[0] = *(const u32x4*)&xb[l16][ks * 32 + kfirst];
        a.q[1] = *(const u32x4*)&xb[l16][ks * 32 + kfirst + 16];
        for (int t = 0; t < 5; ++t) {
            int coln = wave * 80 + t * 16 + l16;
            const unsigned short* wp = &Wb[(size_t)coln * DT_ + ks * 32 + bsel];
            Frag b;
            b.q[0] = *(const u32x4*)(wp);
            b.q[1] = *(const u32x4*)(wp + 16);
            acc[t] = __builtin_amdgcn_wmma_f32_16x16x32_bf16(
                false, a.v, false, b.v, (short)0, acc[t], false, false);
        }
    }
    // direct fragment-layout stores (bias + bf16 convert)
    for (int t = 0; t < 5; ++t) {
        int coln = wave * 80 + t * 16 + l16;
        float bv = bias[coln];
        for (int r = 0; r < 8; ++r)
            Kbf[(size_t)(m0 + rbase + r) * D_ + coln] = f2bf(acc[t][r] + bv);
    }
}

// ---------------- Per-batch attention aggregation ----------------
// scores -> softmax(H) -> sum heads -> query-weighting -> softmax(H) -> aw[B,H]
__global__ __launch_bounds__(256)
void scores_kernel(const unsigned short* __restrict__ Qbf,   // [B,64,640]
                   const unsigned short* __restrict__ Kbf,   // [B,200,640]
                   const float* __restrict__ rnorm,          // [B,64]
                   float* __restrict__ aw_ws,                // [B,200]
                   float* __restrict__ aw_out) {             // [B,200] (d_out tail)
    __shared__ float Ssc[32][208];
    __shared__ float Tsum[32][208];
    __shared__ float nrm[64], qw[64];
    __shared__ float agg[208];
    const int b = blockIdx.x;
    const int tid = threadIdx.x, lane = tid & 31, wave = tid >> 5;
    const int l16 = lane & 15;
    const int kfirst = (lane < 16) ? 0 : 8;
    const int bsel = (lane >> 4) << 3;
    const int rbase = (lane >> 4) << 3;
    const float invscale = 0.039528470752f; // 1/sqrt(640)

    if (tid < 64) nrm[tid] = rnorm[b * 64 + tid];
    if (tid < 208) agg[tid] = 0.f;
    __syncthreads();
    if (tid < 64) {   // query weights: softmax over 64 candidates of ||Qf||
        float mx = -1e30f;
        for (int i = 0; i < 64; ++i) mx = fmaxf(mx, nrm[i]);
        float s = 0.f;
        for (int i = 0; i < 64; ++i) s += __expf(nrm[i] - mx);
        qw[tid] = __expf(nrm[tid] - mx) / s;
    }
    __syncthreads();

    for (int rh = 0; rh < 2; ++rh) {      // 2 halves of 32 candidate rows
        for (int e = tid; e < 32 * 208; e += 256) Tsum[e / 208][e % 208] = 0.f;
        __syncthreads();
        for (int h = 0; h < NH_; ++h) {
            for (int t = wave; t < 26; t += 8) {   // 2 row-tiles x 13 col-tiles
                int rt = t / 13, ct = t % 13;
                v8f acc = (v8f){0.f,0.f,0.f,0.f,0.f,0.f,0.f,0.f};
                for (int ks = 0; ks < 2; ++ks) {
                    int kofs = h * HD_ + ks * 32;
                    Frag a;
                    const unsigned short* qp =
                        &Qbf[(size_t)(b * 64 + rh * 32 + rt * 16 + l16) * D_ + kofs + kfirst];
                    a.q[0] = *(const u32x4*)(qp);
                    a.q[1] = *(const u32x4*)(qp + 16);
                    int col = ct * 16 + l16;
                    int colc = col < H_ ? col : H_ - 1;
                    const unsigned short* kp =
                        &Kbf[(size_t)(b * H_ + colc) * D_ + kofs + bsel];
                    Frag bb;
                    bb.q[0] = *(const u32x4*)(kp);
                    bb.q[1] = *(const u32x4*)(kp + 16);
                    if (col >= H_) {   // per-lane value mask (keeps EXEC all-1 at WMMA)
                        bb.q[0] = (u32x4){0u,0u,0u,0u};
                        bb.q[1] = (u32x4){0u,0u,0u,0u};
                    }
                    acc = __builtin_amdgcn_wmma_f32_16x16x32_bf16(
                        false, a.v, false, bb.v, (short)0, acc, false, false);
                }
                for (int r = 0; r < 8; ++r)
                    Ssc[rt * 16 + rbase + r][ct * 16 + l16] = acc[r] * invscale;
            }
            __syncthreads();
            if (tid < 32) {   // row softmax over 200 keys, accumulate across heads
                float mx = -1e30f;
                for (int m = 0; m < H_; ++m) mx = fmaxf(mx, Ssc[tid][m]);
                float s = 0.f;
                for (int m = 0; m < H_; ++m) s += __expf(Ssc[tid][m] - mx);
                float inv = 1.0f / s;
                for (int m = 0; m < H_; ++m)
                    Tsum[tid][m] += __expf(Ssc[tid][m] - mx) * inv;
            }
            __syncthreads();
        }
        if (tid < H_) {
            float s = 0.f;
            for (int r = 0; r < 32; ++r) s += Tsum[r][tid] * qw[rh * 32 + r];
            agg[tid] += s;
        }
        __syncthreads();
    }
    if (tid < H_) {   // final softmax over 200 clicked docs
        float mx = -1e30f;
        for (int m = 0; m < H_; ++m) mx = fmaxf(mx, agg[m]);
        float s = 0.f;
        for (int m = 0; m < H_; ++m) s += __expf(agg[m] - mx);
        float v = __expf(agg[tid] - mx) / s;
        aw_ws[b * H_ + tid] = v;
        aw_out[b * H_ + tid] = v;
    }
}

// ---------------- Gate GEMM fused with sigmoid/residual/LayerNorm ----------------
__global__ __launch_bounds__(256)
void gate_kernel(const float* __restrict__ CN,               // clicked_news [B*H,640]
                 const unsigned short* __restrict__ Wgb,     // [640,640] bf16
                 const float* __restrict__ bg,
                 const float* __restrict__ aw,               // [B*H]
                 const float* __restrict__ gamma,
                 const float* __restrict__ beta,
                 float* __restrict__ out) {
    __shared__ unsigned short wcb[16][D_];   // 20 KB
    __shared__ float accb[16][D_];           // 40 KB
    __shared__ float awl[16];
    __shared__ float redS[16][16], redQ[16][16];
    __shared__ float muA[16], rsA[16];
    const int m0 = blockIdx.x * 16;
    const int tid = threadIdx.x, lane = tid & 31, wave = tid >> 5;
    const int l16 = lane & 15;
    const int kfirst = (lane < 16) ? 0 : 8;
    const int bsel = (lane >> 4) << 3;
    const int rbase = (lane >> 4) << 3;

    if (tid < 16) awl[tid] = aw[m0 + tid];
    __syncthreads();
    for (int e = tid; e < 16 * D_; e += 256) {   // weighted clicked -> bf16 LDS
        int r = e / D_, c = e - r * D_;
        wcb[r][c] = f2bf(awl[r] * CN[(size_t)(m0 + r) * D_ + c]);
    }
    __syncthreads();

    v8f acc[5];
    for (int t = 0; t < 5; ++t) acc[t] = (v8f){0.f,0.f,0.f,0.f,0.f,0.f,0.f,0.f};
    for (int ks = 0; ks < D_ / 32; ++ks) {
        Frag a;
        a.q[0] = *(const u32x4*)&wcb[l16][ks * 32 + kfirst];
        a.q[1] = *(const u32x4*)&wcb[l16][ks * 32 + kfirst + 16];
        for (int t = 0; t < 5; ++t) {
            int coln = wave * 80 + t * 16 + l16;
            const unsigned short* wp = &Wgb[(size_t)coln * D_ + ks * 32 + bsel];
            Frag b;
            b.q[0] = *(const u32x4*)(wp);
            b.q[1] = *(const u32x4*)(wp + 16);
            acc[t] = __builtin_amdgcn_wmma_f32_16x16x32_bf16(
                false, a.v, false, b.v, (short)0, acc[t], false, false);
        }
    }
    for (int t = 0; t < 5; ++t)
        for (int r = 0; r < 8; ++r)
            accb[rbase + r][wave * 80 + t * 16 + l16] = acc[t][r];
    __syncthreads();

    {   // sigmoid gate + residual mix; LN partial sums
        int r = tid >> 4, seg = tid & 15;
        float s = 0.f, sq = 0.f;
        float a_ = awl[r];
        for (int c = seg * 40; c < seg * 40 + 40; ++c) {
            float cn = CN[(size_t)(m0 + r) * D_ + c];
            float wc = a_ * cn;
            float gi = accb[r][c] + bg[c];
            float g = 1.0f / (1.0f + __expf(-gi));
            float o = g * wc + (1.0f - g) * cn;
            accb[r][c] = o;
            s += o; sq += o * o;
        }
        redS[r][seg] = s; redQ[r][seg] = sq;
    }
    __syncthreads();
    if ((tid & 15) == 0) {
        int r = tid >> 4;
        float s = 0.f, sq = 0.f;
        for (int i = 0; i < 16; ++i) { s += redS[r][i]; sq += redQ[r][i]; }
        float mu = s / (float)D_;
        float var = sq / (float)D_ - mu * mu;
        muA[r] = mu;
        rsA[r] = rsqrtf(var + 1e-5f);
    }
    __syncthreads();
    {   // normalized write-out (single pass over the 131 MB output)
        int r = tid >> 4, seg = tid & 15;
        float mu = muA[r], rs = rsA[r];
        for (int c = seg * 40; c < seg * 40 + 40; ++c)
            out[(size_t)(m0 + r) * D_ + c] = (accb[r][c] - mu) * rs * gamma[c] + beta[c];
    }
}

extern "C" void kernel_launch(void* const* d_in, const int* in_sizes, int n_in,
                              void* d_out, int out_size, void* d_ws, size_t ws_size,
                              hipStream_t stream) {
    const float* clicked_news   = (const float*)d_in[0];
    const float* clicked_topics = (const float*)d_in[1];
    const float* cand_topics    = (const float*)d_in[2];
    const float* Wq = (const float*)d_in[3];
    const float* bq = (const float*)d_in[4];
    const float* Wk = (const float*)d_in[5];
    const float* bk = (const float*)d_in[6];
    // d_in[7]=Wv, d_in[8]=bv: dead code w.r.t. outputs
    const float* Wg = (const float*)d_in[9];
    const float* bg = (const float*)d_in[10];
    const float* gamma = (const float*)d_in[11];
    const float* beta  = (const float*)d_in[12];
    float* out = (float*)d_out;

    char* ws = (char*)d_ws;
    size_t off = 0;
    auto alloc = [&](size_t bytes) -> void* {
        void* p = ws + off;
        off = (off + bytes + 255) & ~(size_t)255;
        return p;
    };
    unsigned short* Wq_bf = (unsigned short*)alloc((size_t)D_ * DT_ * 2);
    unsigned short* Wk_bf = (unsigned short*)alloc((size_t)D_ * DT_ * 2);
    unsigned short* Wg_bf = (unsigned short*)alloc((size_t)D_ * D_ * 2);
    unsigned short* Q_bf  = (unsigned short*)alloc((size_t)B_ * N_ * D_ * 2);
    unsigned short* K_bf  = (unsigned short*)alloc((size_t)B_ * H_ * D_ * 2);
    float* rnorm = (float*)alloc((size_t)B_ * N_ * 4);
    float* aw_ws = (float*)alloc((size_t)B_ * H_ * 4);

    cvt_f32_bf16<<<(D_ * DT_ + 255) / 256, 256, 0, stream>>>(Wq, Wq_bf, D_ * DT_);
    cvt_f32_bf16<<<(D_ * DT_ + 255) / 256, 256, 0, stream>>>(Wk, Wk_bf, D_ * DT_);
    cvt_f32_bf16<<<(D_ * D_ + 255) / 256, 256, 0, stream>>>(Wg, Wg_bf, D_ * D_);

    qproj_kernel<<<(B_ * N_) / 16, 256, 0, stream>>>(cand_topics, Wq_bf, bq, Q_bf, rnorm);
    kproj_kernel<<<(B_ * H_) / 16, 256, 0, stream>>>(clicked_topics, Wk_bf, bk, K_bf);

    float* aw_out = out + (size_t)B_ * H_ * D_;   // second tuple output [B,H]
    scores_kernel<<<B_, 256, 0, stream>>>(Q_bf, K_bf, rnorm, aw_ws, aw_out);

    gate_kernel<<<(B_ * H_) / 16, 256, 0, stream>>>(clicked_news, Wg_bf, bg, aw_ws,
                                                    gamma, beta, out);
}
